// ChaRNN_17910013624559
// MI455X (gfx1250) — compile-verified, matrix operand order checked by reading, not run on hardware
//
#include <hip/hip_runtime.h>
#include <math.h>

#define H_      1024
#define F_      512
#define B_      64
#define T_      512
#define G4_     (4 * H_)
#define KTOT    (F_ + H_)     // 1536 (x-part then h-part)
#define NCHUNK  (KTOT / 32)   // 48 K-chunks of 32
#define XCHUNK  (F_ / 32)     // 16
#define HCHUNK  (H_ / 32)     // 32
#define NWG     256           // 64 n-tiles x 4 m-tiles, persistent grid

typedef __attribute__((ext_vector_type(8)))  float  v8f;
typedef __attribute__((ext_vector_type(8)))  __bf16 v8bf;
typedef __attribute__((ext_vector_type(16))) __bf16 v16bf;

// Fragment-packed block: 16 (rows/cols) x 32 (K) bf16 = 512 halves = 1KB.
// Element (n, K = h2*16 + kp*8 + r) lives at ((h2*2 + kp)*16 + n)*8 + r, so
// lane l (n = l&15, kp = l>>4) reads two coalesced b128s at l*16B and 512B + l*16B,
// landing exactly in the ISA's 16-bit A(16x32)/B(32x16) VGPR layout.
__device__ __forceinline__ v16bf ldfrag(const __bf16* __restrict__ blk, int lane) {
    const v8bf a = *(const v8bf*)(blk + lane * 8);
    const v8bf b = *(const v8bf*)(blk + 256 + lane * 8);
    return __builtin_shufflevector(a, b, 0, 1, 2, 3, 4, 5, 6, 7,
                                         8, 9, 10, 11, 12, 13, 14, 15);
}

__device__ __forceinline__ v8f wmma_bf16(v16bf a, v16bf b, v8f c) {
    return __builtin_amdgcn_wmma_f32_16x16x32_bf16(
        false, a, false, b, (short)0, c, false, false);
}

// ---- Pack [W;U]^T (4096 x 1536) into bf16 hi/lo fragment blocks -----------
__global__ __launch_bounds__(256)
void pack_weights(const float* __restrict__ W, const float* __restrict__ U,
                  unsigned short* __restrict__ wp_hi_u,
                  unsigned short* __restrict__ wp_lo_u) {
    __bf16* wp_hi = (__bf16*)wp_hi_u;
    __bf16* wp_lo = (__bf16*)wp_lo_u;
    const int idx = blockIdx.x * 256 + threadIdx.x;   // < 256*NCHUNK*512
    const int gt  = idx / (NCHUNK * 512);
    const int rem = idx - gt * (NCHUNK * 512);
    const int kc  = rem >> 9;
    const int p   = rem & 511;
    const int q   = p >> 7;            // h2*2 + kp
    const int n   = (p >> 3) & 15;
    const int r   = p & 7;
    const int k   = kc * 32 + (q >> 1) * 16 + (q & 1) * 8 + r;
    const int g   = gt * 16 + n;
    const float v = (k < F_) ? W[(size_t)k * G4_ + g]
                             : U[(size_t)(k - F_) * G4_ + g];
    const __bf16 hi = (__bf16)v;
    wp_hi[idx] = hi;
    wp_lo[idx] = (__bf16)(v - (float)hi);
}

// ---- Pack x (B,T,F) into bf16 hi/lo A-fragment blocks ---------------------
__global__ __launch_bounds__(256)
void pack_x(const float* __restrict__ x,
            unsigned short* __restrict__ xp_hi_u,
            unsigned short* __restrict__ xp_lo_u) {
    __bf16* xp_hi = (__bf16*)xp_hi_u;
    __bf16* xp_lo = (__bf16*)xp_lo_u;
    const int idx   = blockIdx.x * 256 + threadIdx.x;  // < 4*T*XCHUNK*512
    const int btile = idx >> 22;
    const int rem   = idx & 4194303;
    const int t     = rem >> 13;
    const int rem2  = rem & 8191;
    const int kc    = rem2 >> 9;
    const int p     = rem2 & 511;
    const int q     = p >> 7;
    const int m     = (p >> 3) & 15;
    const int r     = p & 7;
    const int f     = kc * 32 + (q >> 1) * 16 + (q & 1) * 8 + r;
    const int b     = btile * 16 + m;
    const float v   = x[((size_t)b * T_ + t) * F_ + f];
    const __bf16 hi = (__bf16)v;
    xp_hi[idx] = hi;
    xp_lo[idx] = (__bf16)(v - (float)hi);
}

__global__ void zero_kernel(float* __restrict__ p, int n) {
    const int i = blockIdx.x * blockDim.x + threadIdx.x;
    if (i < n) p[i] = 0.0f;
}

// ---- Persistent LSTM: all 512 timesteps in one kernel ---------------------
// Grid = 256 WGs (64 n-tiles x 4 m-tiles), block = 128 = 4 waves (one per gate).
// Device-wide barrier between steps via agent-scope atomic + release/acquire
// fences; weights stay L2-resident across the whole sequence.
__global__ __launch_bounds__(128)
void lstm_persistent(const unsigned short* __restrict__ xp_hi_u,
                     const unsigned short* __restrict__ xp_lo_u,
                     const unsigned short* __restrict__ wp_hi_u,
                     const unsigned short* __restrict__ wp_lo_u,
                     unsigned short* __restrict__ hp_hi_u,   // [2][B*H] ping-pong
                     unsigned short* __restrict__ hp_lo_u,   // [2][B*H]
                     float* __restrict__ cb,                 // [2][B*H]
                     const float* __restrict__ bias,
                     float* __restrict__ out,
                     float* __restrict__ h_last, float* __restrict__ c_last,
                     unsigned int* __restrict__ sync)
{
    __shared__ float zs[4][16][17];

    const size_t S  = (size_t)B_ * H_;
    const int lane  = threadIdx.x & 31;
    const int wave  = threadIdx.x >> 5;      // gate: 0=i 1=f 2=g 3=o
    const int wg    = blockIdx.x;
    const int ntile = wg & 63;               // 0..63 over H
    const int mtile = wg >> 6;               // 0..3  over B
    const int gt    = wave * 64 + ntile;
    const int col   = lane & 15;
    const int kp    = lane >> 4;

    const __bf16* wbh = (const __bf16*)wp_hi_u + (size_t)gt * NCHUNK * 512;
    const __bf16* wbl = (const __bf16*)wp_lo_u + (size_t)gt * NCHUNK * 512;

    for (int t = 0; t < T_; ++t) {
        const int cur = t & 1, nxt = cur ^ 1;
        const __bf16* xbh = (const __bf16*)xp_hi_u + ((size_t)(mtile * T_ + t) * XCHUNK) * 512;
        const __bf16* xbl = (const __bf16*)xp_lo_u + ((size_t)(mtile * T_ + t) * XCHUNK) * 512;
        const __bf16* hbh = (const __bf16*)hp_hi_u + (size_t)cur * S + (size_t)mtile * HCHUNK * 512;
        const __bf16* hbl = (const __bf16*)hp_lo_u + (size_t)cur * S + (size_t)mtile * HCHUNK * 512;
        __bf16* hnh = (__bf16*)hp_hi_u + (size_t)nxt * S;
        __bf16* hnl = (__bf16*)hp_lo_u + (size_t)nxt * S;
        const float* c_prev = cb + (size_t)cur * S;
        float*       c_next = cb + (size_t)nxt * S;

        v8f acc = {};

        // ---- x_t @ W : 16 chunks of K=32, split-bf16 fp32 emulation ----
        #pragma unroll 4
        for (int kc = 0; kc < XCHUNK; ++kc) {
            const v16bf ah = ldfrag(xbh + (size_t)kc * 512, lane);
            const v16bf al = ldfrag(xbl + (size_t)kc * 512, lane);
            const v16bf bh = ldfrag(wbh + (size_t)kc * 512, lane);
            const v16bf bl = ldfrag(wbl + (size_t)kc * 512, lane);
            acc = wmma_bf16(ah, bh, acc);
            acc = wmma_bf16(ah, bl, acc);
            acc = wmma_bf16(al, bh, acc);
        }

        // ---- h_prev @ U : 32 chunks of K=32 (weight chunks 16..47) ----
        #pragma unroll 4
        for (int kc = 0; kc < HCHUNK; ++kc) {
            const v16bf ah = ldfrag(hbh + (size_t)kc * 512, lane);
            const v16bf al = ldfrag(hbl + (size_t)kc * 512, lane);
            const v16bf bh = ldfrag(wbh + (size_t)(XCHUNK + kc) * 512, lane);
            const v16bf bl = ldfrag(wbl + (size_t)(XCHUNK + kc) * 512, lane);
            acc = wmma_bf16(ah, bh, acc);
            acc = wmma_bf16(ah, bl, acc);
            acc = wmma_bf16(al, bh, acc);
        }

        // Spill z-tile. C/D layout: VGPR r holds (M = r + 8*kp, N = col).
        #pragma unroll
        for (int r = 0; r < 8; ++r)
            zs[wave][r + 8 * kp][col] = acc[r];

        __syncthreads();

        // Gate math: 256 tile elements, 128 threads -> 2 each.
        for (int e = threadIdx.x; e < 256; e += 128) {
            const int m    = e >> 4;
            const int n    = e & 15;
            const int brow = mtile * 16 + m;
            const int j    = ntile * 16 + n;

            const float zi = zs[0][m][n] + bias[j];
            const float zf = zs[1][m][n] + bias[H_ + j];
            const float zg = zs[2][m][n] + bias[2 * H_ + j];
            const float zo = zs[3][m][n] + bias[3 * H_ + j];

            const float ig = 1.0f / (1.0f + expf(-zi));
            const float fg = 1.0f / (1.0f + expf(-zf));
            const float gg = tanhf(zg);
            const float og = 1.0f / (1.0f + expf(-zo));

            const float cp = c_prev[(size_t)brow * H_ + j];
            const float cn = fg * cp + ig * gg;
            const float hn = og * tanhf(cn);

            c_next[(size_t)brow * H_ + j] = cn;
            out[((size_t)brow * T_ + t) * H_ + j] = hn;

            // Re-split h into the packed A-fragment layout for the next step.
            const int kc   = j >> 5;
            const int remj = j & 31;               // h2*16 + kp*8 + r
            const int q    = remj >> 3;            // h2*2 + kp
            const int r    = remj & 7;
            const size_t hidx = ((size_t)mtile * HCHUNK + kc) * 512
                              + (size_t)(q * 16 + m) * 8 + r;
            const __bf16 hh = (__bf16)hn;
            hnh[hidx] = hh;
            hnl[hidx] = (__bf16)(hn - (float)hh);

            if (t == T_ - 1) {
                h_last[(size_t)brow * H_ + j] = hn;
                c_last[(size_t)brow * H_ + j] = cn;
            }
        }

        // ---- device-wide step barrier (skip after the final step) ----
        if (t < T_ - 1) {
            __builtin_amdgcn_fence(__ATOMIC_RELEASE, "agent");  // publish h/c stores
            __syncthreads();                                    // whole WG published
            if (threadIdx.x == 0) {
                __hip_atomic_fetch_add(sync, 1u, __ATOMIC_RELAXED,
                                       __HIP_MEMORY_SCOPE_AGENT);
                const unsigned target = (unsigned)NWG * (unsigned)(t + 1);
                while (__hip_atomic_load(sync, __ATOMIC_RELAXED,
                                         __HIP_MEMORY_SCOPE_AGENT) < target)
                    __builtin_amdgcn_s_sleep(2);
            }
            __syncthreads();
            __builtin_amdgcn_fence(__ATOMIC_ACQUIRE, "agent");  // see others' h/c
        }
    }
}

extern "C" void kernel_launch(void* const* d_in, const int* in_sizes, int n_in,
                              void* d_out, int out_size, void* d_ws, size_t ws_size,
                              hipStream_t stream) {
    (void)in_sizes; (void)n_in; (void)out_size; (void)ws_size;

    const float* x    = (const float*)d_in[0];  // (B,T,F)
    const float* W    = (const float*)d_in[1];  // (F,4H)
    const float* U    = (const float*)d_in[2];  // (H,4H)
    const float* bias = (const float*)d_in[3];  // (4H)

    float* out    = (float*)d_out;
    float* h_last = out + (size_t)B_ * T_ * H_;
    float* c_last = h_last + (size_t)B_ * H_;

    // ---- workspace layout ----
    char* ws = (char*)d_ws;
    const size_t WP_HALVES = (size_t)256 * NCHUNK * 512;        // 6,291,456
    const size_t XP_HALVES = (size_t)4 * T_ * XCHUNK * 512;     // 16,777,216
    const size_t S         = (size_t)B_ * H_;                   // 65,536

    unsigned short* wp_hi = (unsigned short*)ws;
    unsigned short* wp_lo = wp_hi + WP_HALVES;
    unsigned short* xp_hi = wp_lo + WP_HALVES;
    unsigned short* xp_lo = xp_hi + XP_HALVES;
    unsigned short* hp_hi = xp_lo + XP_HALVES;   // [2][S]
    unsigned short* hp_lo = hp_hi + 2 * S;       // [2][S]
    float*          cb    = (float*)(hp_lo + 2 * S);            // [2][S]
    unsigned int*   sync  = (unsigned int*)(cb + 2 * S);

    // ---- per-launch packing passes ----
    pack_weights<<<(int)(WP_HALVES / 256), 256, 0, stream>>>(W, U, wp_hi, wp_lo);
    pack_x<<<(int)(XP_HALVES / 256), 256, 0, stream>>>(x, xp_hi, xp_lo);
    // zero: hp (4*S halves = 2*S floats) + cb (2*S floats) + sync (1)
    const int zn = (int)(4 * S) + 1;
    zero_kernel<<<(zn + 255) / 256, 256, 0, stream>>>((float*)hp_hi, zn);

    // ---- one persistent kernel runs all 512 timesteps ----
    lstm_persistent<<<NWG, 128, 0, stream>>>(
        xp_hi, xp_lo, wp_hi, wp_lo,
        hp_hi, hp_lo, cb,
        bias, out, h_last, c_last, sync);
}
